// MotionTransformer_31379031064722
// MI455X (gfx1250) — compile-verified, matrix operand order checked by reading, not run on hardware
//
#include <hip/hip_runtime.h>
#include <hip/hip_bf16.h>

// ---------------------------------------------------------------------------
// MotionTransformer forward for MI455X (gfx1250, wave32, WMMA).
// - All large GEMMs: bf16 WMMA (v_wmma_f32_16x16x32_bf16, fp32 accumulate).
// - Weights converted fp32->bf16 AND pre-transposed to [N,K] each call, so both
//   A and B LDS tiles stage as contiguous 16B chunks (no LDS scatter).
// - GEMM staging uses double-buffered GLOBAL_LOAD_ASYNC_TO_LDS_B128 (ASYNCcnt),
//   overlapping HBM fetch with WMMA issue; one barrier per K-step.
// - Linear-attention core (64x64 state per (b,head)) is VALU+LDS (<2% of FLOPs).
// ---------------------------------------------------------------------------

typedef __bf16 bf16;
typedef bf16  v16bf __attribute__((ext_vector_type(16)));
typedef bf16  v8bf  __attribute__((ext_vector_type(8)));
typedef float v8f   __attribute__((ext_vector_type(8)));

// Model constants
#define LAYERS 8
#define BATCH  64
#define TT     240
#define DIM    512
#define AUD    256
#define NHEAD  8
#define TEMB   2048
#define FFN    1024
#define ROWS   (BATCH * TT)          // 15360
#define QKVN   (3 * DIM)             // 1536

// ---------------------------------------------------------------------------
// CDNA5 async global->LDS copy (16B per lane), tracked by ASYNCcnt
// ---------------------------------------------------------------------------
__device__ __forceinline__ void async_copy16(unsigned lds_off, const bf16* gaddr) {
    asm volatile("global_load_async_to_lds_b128 %0, %1, off"
                 :: "v"(lds_off), "v"(gaddr)
                 : "memory");
}

__device__ __forceinline__ void wait_async0() {
#if __has_builtin(__builtin_amdgcn_s_wait_asynccnt)
    __builtin_amdgcn_s_wait_asynccnt(0);
#else
    asm volatile("s_wait_asynccnt 0x0" ::: "memory");
#endif
}

// ---------------------------------------------------------------------------
// Elementwise conversion kernels
// ---------------------------------------------------------------------------

// transpose-convert: src [layers, Kd, Nd] fp32 -> dst [layers, Nd, Kd] bf16
__global__ void cvt_t_bf16_kernel(const float* __restrict__ s, bf16* __restrict__ d,
                                  int Kd, int Nd, long n) {
    long i = (long)blockIdx.x * blockDim.x + threadIdx.x;
    if (i >= n) return;
    long per = (long)Kd * Nd;
    long l = i / per;
    long r = i - l * per;
    int nn = (int)(r / Kd);
    int kk = (int)(r - (long)nn * Kd);
    d[i] = (bf16)s[l * per + (long)kk * Nd + nn];
}

__global__ void silu_cvt_kernel(const float* __restrict__ s, bf16* __restrict__ d, long n) {
    long i = (long)blockIdx.x * blockDim.x + threadIdx.x;
    if (i < n) {
        float x = s[i];
        d[i] = (bf16)(x / (1.0f + __expf(-x)));
    }
}

// pack+transpose q/k/v weights [L,512,512] each -> [L,1536,512] bf16 ([N,K])
__global__ void pack_qkv_w_kernel(const float* __restrict__ qw, const float* __restrict__ kw,
                                  const float* __restrict__ vw, bf16* __restrict__ dst) {
    long n = (long)LAYERS * QKVN * DIM;
    long i = (long)blockIdx.x * blockDim.x + threadIdx.x;
    if (i >= n) return;
    long l = i / ((long)QKVN * DIM);
    long r = i - l * (long)QKVN * DIM;
    int  c = (int)(r / DIM);            // output row (0..1535)
    int  k = (int)(r - (long)c * DIM);  // output col = K index
    const float* src = (c < DIM) ? qw : (c < 2 * DIM) ? kw : vw;
    int cc = c & (DIM - 1);
    dst[i] = (bf16)src[((long)l * DIM + k) * DIM + cc];
}

__global__ void pack_qkv_b_kernel(const float* __restrict__ qb, const float* __restrict__ kb,
                                  const float* __restrict__ vb, float* __restrict__ dst) {
    int n = LAYERS * QKVN;
    int i = blockIdx.x * blockDim.x + threadIdx.x;
    if (i >= n) return;
    int l = i / QKVN;
    int c = i - l * QKVN;
    const float* src = (c < DIM) ? qb : (c < 2 * DIM) ? kb : vb;
    dst[i] = src[l * DIM + (c & (DIM - 1))];
}

// concat(h fp32 [R,512], xf fp32 [R,256]) -> bf16 [R,768]
__global__ void concat_bf16_kernel(const float* __restrict__ h, const float* __restrict__ xf,
                                   bf16* __restrict__ out, long n) {
    long i = (long)blockIdx.x * blockDim.x + threadIdx.x;
    if (i >= n) return;
    long r = i / (DIM + AUD);
    int  c = (int)(i - r * (DIM + AUD));
    float v = (c < DIM) ? h[r * DIM + c] : xf[r * AUD + (c - DIM)];
    out[i] = (bf16)v;
}

// ---------------------------------------------------------------------------
// LayerNorm (C=512) -> bf16
// ---------------------------------------------------------------------------
__global__ __launch_bounds__(256) void ln512_bf16_kernel(const float* __restrict__ x,
        const float* __restrict__ g, const float* __restrict__ b, bf16* __restrict__ out) {
    int row = blockIdx.x, tid = threadIdx.x;
    __shared__ float red[256];
    const float* xr = x + (long)row * DIM;
    float v0 = xr[tid], v1 = xr[tid + 256];
    red[tid] = v0 + v1; __syncthreads();
    for (int s = 128; s > 0; s >>= 1) { if (tid < s) red[tid] += red[tid + s]; __syncthreads(); }
    float mean = red[0] * (1.0f / DIM);
    __syncthreads();
    float d0 = v0 - mean, d1 = v1 - mean;
    red[tid] = d0 * d0 + d1 * d1; __syncthreads();
    for (int s = 128; s > 0; s >>= 1) { if (tid < s) red[tid] += red[tid + s]; __syncthreads(); }
    float rstd = rsqrtf(red[0] * (1.0f / DIM) + 1e-5f);
    out[(long)row * DIM + tid]       = (bf16)(d0 * rstd * g[tid] + b[tid]);
    out[(long)row * DIM + tid + 256] = (bf16)(d1 * rstd * g[tid + 256] + b[tid + 256]);
}

// AdaLN + SiLU -> bf16:  silu( LN(y)*(1+scale_b) + shift_b ),  e = [B,1024] (scale|shift)
__global__ __launch_bounds__(256) void adaln_silu_kernel(const float* __restrict__ y,
        const float* __restrict__ e, const float* __restrict__ g, const float* __restrict__ b,
        bf16* __restrict__ out) {
    int row = blockIdx.x, tid = threadIdx.x;
    int bb = row / TT;
    __shared__ float red[256];
    const float* xr = y + (long)row * DIM;
    float v0 = xr[tid], v1 = xr[tid + 256];
    red[tid] = v0 + v1; __syncthreads();
    for (int s = 128; s > 0; s >>= 1) { if (tid < s) red[tid] += red[tid + s]; __syncthreads(); }
    float mean = red[0] * (1.0f / DIM);
    __syncthreads();
    float d0 = v0 - mean, d1 = v1 - mean;
    red[tid] = d0 * d0 + d1 * d1; __syncthreads();
    for (int s = 128; s > 0; s >>= 1) { if (tid < s) red[tid] += red[tid + s]; __syncthreads(); }
    float rstd = rsqrtf(red[0] * (1.0f / DIM) + 1e-5f);
    const float* eb = e + (long)bb * (2 * DIM);
    #pragma unroll
    for (int half = 0; half < 2; ++half) {
        int c = tid + half * 256;
        float d = half ? d1 : d0;
        float hv = (d * rstd * g[c] + b[c]) * (1.0f + eb[c]) + eb[DIM + c];
        out[(long)row * DIM + c] = (bf16)(hv / (1.0f + __expf(-hv)));
    }
}

// ---------------------------------------------------------------------------
// Tiled bf16 WMMA GEMM:  C[M,N] = act(A[M,K] @ W[K,N] + bias [+ resid])
// W is pre-transposed in memory as Bw[N,K]. Block tile 128x128, BK=32, 8 waves,
// each wave 32x64 (2x4 WMMA tiles). Double-buffered async global->LDS staging.
// ---------------------------------------------------------------------------
__global__ __launch_bounds__(256) void gemm_bf16_kernel(
        const bf16* __restrict__ A, const bf16* __restrict__ Bw,
        const float* __restrict__ bias, const float* __restrict__ resid,
        float* __restrict__ Cf, bf16* __restrict__ Cb,
        int M, int N, int K, int act) {
    __shared__ __align__(16) bf16 sA[2][128][40];   // [m][k], +8 pad
    __shared__ __align__(16) bf16 sB[2][128][40];   // [n][k], +8 pad

    int tid = threadIdx.x;
    int bn0 = blockIdx.x * 128;
    int bm0 = blockIdx.y * 128;
    int wid = tid >> 5, lane = tid & 31;
    int wm = wid & 3, wn = wid >> 2;           // 4 wave-rows x 2 wave-cols
    int m16 = lane & 15, khalf = lane >> 4;

    // staging chunk coords: each thread copies 2 x 16B of A and of B per tile
    int r0 = tid >> 2, kc0 = (tid & 3) << 3;   // rows 0..63
    int r1 = r0 + 64;                           // rows 64..127, same kc

    v8f acc[2][4];
    #pragma unroll
    for (int i = 0; i < 2; ++i)
        #pragma unroll
        for (int j = 0; j < 4; ++j)
            #pragma unroll
            for (int v = 0; v < 8; ++v) acc[i][j][v] = 0.0f;

    auto stage = [&](int kb, int buf) {
        long ko = (long)kb * 32 + kc0;
        if (bm0 + r0 < M)
            async_copy16((unsigned)(size_t)&sA[buf][r0][kc0], A + (long)(bm0 + r0) * K + ko);
        if (bm0 + r1 < M)
            async_copy16((unsigned)(size_t)&sA[buf][r1][kc0], A + (long)(bm0 + r1) * K + ko);
        async_copy16((unsigned)(size_t)&sB[buf][r0][kc0], Bw + (long)(bn0 + r0) * K + ko);
        async_copy16((unsigned)(size_t)&sB[buf][r1][kc0], Bw + (long)(bn0 + r1) * K + ko);
    };

    int nkb = K >> 5;
    stage(0, 0);
    for (int kb = 0; kb < nkb; ++kb) {
        int cur = kb & 1;
        wait_async0();          // this wave's chunks of tile kb have landed
        __syncthreads();        // all waves' chunks have landed; prev compute done
        if (kb + 1 < nkb) stage(kb + 1, cur ^ 1);

        v16bf af[2], bfm[4];
        #pragma unroll
        for (int mt = 0; mt < 2; ++mt) {
            int r = wm * 32 + mt * 16 + m16;
            v8bf lo = *(const v8bf*)&sA[cur][r][khalf * 8];
            v8bf hi = *(const v8bf*)&sA[cur][r][16 + khalf * 8];
            af[mt] = __builtin_shufflevector(lo, hi, 0,1,2,3,4,5,6,7,8,9,10,11,12,13,14,15);
        }
        #pragma unroll
        for (int nt = 0; nt < 4; ++nt) {
            int cix = wn * 64 + nt * 16 + m16;
            v8bf lo = *(const v8bf*)&sB[cur][cix][khalf * 8];
            v8bf hi = *(const v8bf*)&sB[cur][cix][16 + khalf * 8];
            bfm[nt] = __builtin_shufflevector(lo, hi, 0,1,2,3,4,5,6,7,8,9,10,11,12,13,14,15);
        }
        #pragma unroll
        for (int mt = 0; mt < 2; ++mt)
            #pragma unroll
            for (int nt = 0; nt < 4; ++nt)
                acc[mt][nt] = __builtin_amdgcn_wmma_f32_16x16x32_bf16(
                    false, af[mt], false, bfm[nt], (short)0, acc[mt][nt], false, false);
        // no trailing barrier: next iteration's top barrier orders reuse of this buffer
    }

    // epilogue: lane holds (N=lane%16, M=vgpr + 8*(lane/16)) of each 16x16 tile
    int n16 = lane & 15, mhalf = lane >> 4;
    #pragma unroll
    for (int mt = 0; mt < 2; ++mt)
        #pragma unroll
        for (int nt = 0; nt < 4; ++nt) {
            int gcol = bn0 + wn * 64 + nt * 16 + n16;
            float bv = bias ? bias[gcol] : 0.0f;
            #pragma unroll
            for (int v = 0; v < 8; ++v) {
                int grow = bm0 + wm * 32 + mt * 16 + mhalf * 8 + v;
                if (grow < M) {
                    float val = acc[mt][nt][v] + bv;
                    if (resid) val += resid[(long)grow * N + gcol];
                    if (act == 1) val = 0.5f * val * (1.0f + erff(val * 0.70710678f)); // exact GELU
                    if (Cf) Cf[(long)grow * N + gcol] = val;
                    if (Cb) Cb[(long)grow * N + gcol] = (bf16)val;
                }
            }
        }
}

// ---------------------------------------------------------------------------
// Linear attention: one block per (b, head). qkv fp32 [ROWS,1536] packed q|k|v.
//   k-softmax over T (masked), v*mask, state = k^T v (64x64), y = softmax_d(q) @ state
// ---------------------------------------------------------------------------
__global__ __launch_bounds__(256) void linattn_kernel(const float* __restrict__ qkv,
        const float* __restrict__ mask, float* __restrict__ y) {
    int bh = blockIdx.x;
    int b = bh >> 3, hd = bh & 7;
    int tid = threadIdx.x;

    __shared__ float s_state[64][65];
    __shared__ float s_k[16][64];
    __shared__ float s_v[16][64];
    __shared__ float s_q[32][64];
    __shared__ float s_red[4][64];
    __shared__ float s_cmax[64];
    __shared__ float s_csum[64];

    const int KO = DIM, VO = 2 * DIM;
    int d = tid & 63, part = tid >> 6;

    // column max of masked k over T
    float mx = -3.0e38f;
    for (int t = part; t < TT; t += 4) {
        float mval = mask[b * TT + t];
        float kv = qkv[((long)(b * TT + t)) * QKVN + KO + hd * 64 + d] + (1.0f - mval) * (-1.0e6f);
        mx = fmaxf(mx, kv);
    }
    s_red[part][d] = mx; __syncthreads();
    if (part == 0)
        s_cmax[d] = fmaxf(fmaxf(s_red[0][d], s_red[1][d]), fmaxf(s_red[2][d], s_red[3][d]));
    __syncthreads();
    // column sum of exp
    float cm = s_cmax[d];
    float sm = 0.0f;
    for (int t = part; t < TT; t += 4) {
        float mval = mask[b * TT + t];
        float kv = qkv[((long)(b * TT + t)) * QKVN + KO + hd * 64 + d] + (1.0f - mval) * (-1.0e6f);
        sm += __expf(kv - cm);
    }
    s_red[part][d] = sm; __syncthreads();
    if (part == 0)
        s_csum[d] = s_red[0][d] + s_red[1][d] + s_red[2][d] + s_red[3][d];
    __syncthreads();

    // state[d][l] = sum_t softk[t,d] * v[t,l], register-accumulated, t tiled via LDS
    int d0 = tid >> 2, l0 = (tid & 3) << 4;
    float acc[16];
    #pragma unroll
    for (int j = 0; j < 16; ++j) acc[j] = 0.0f;
    for (int t0 = 0; t0 < TT; t0 += 16) {
        #pragma unroll
        for (int i = 0; i < 4; ++i) {
            int idx = tid * 4 + i;
            int tt = idx >> 6, cc = idx & 63;
            int t = t0 + tt;
            float mval = mask[b * TT + t];
            long base = ((long)(b * TT + t)) * QKVN;
            float kv = qkv[base + KO + hd * 64 + cc] + (1.0f - mval) * (-1.0e6f);
            s_k[tt][cc] = __expf(kv - s_cmax[cc]) / s_csum[cc];
            s_v[tt][cc] = qkv[base + VO + hd * 64 + cc] * mval;
        }
        __syncthreads();
        #pragma unroll
        for (int tt = 0; tt < 16; ++tt) {
            float kd = s_k[tt][d0];
            #pragma unroll
            for (int j = 0; j < 16; ++j) acc[j] += kd * s_v[tt][l0 + j];
        }
        __syncthreads();
    }
    #pragma unroll
    for (int j = 0; j < 16; ++j) s_state[d0][l0 + j] = acc[j];
    __syncthreads();

    // y[t,l] = softmax_d(q[t,:]) . state[:,l]
    for (int t0 = 0; t0 < TT; t0 += 32) {
        if (tid < 32) {
            int t = t0 + tid;
            if (t < TT) {
                long base = ((long)(b * TT + t)) * QKVN + hd * 64;
                float qm = -3.0e38f;
                for (int dd = 0; dd < 64; ++dd) qm = fmaxf(qm, qkv[base + dd]);
                float qs = 0.0f;
                for (int dd = 0; dd < 64; ++dd) {
                    float e = __expf(qkv[base + dd] - qm);
                    s_q[tid][dd] = e; qs += e;
                }
                float inv = 1.0f / qs;
                for (int dd = 0; dd < 64; ++dd) s_q[tid][dd] *= inv;
            }
        }
        __syncthreads();
        #pragma unroll
        for (int i = 0; i < 8; ++i) {
            int idx = tid * 8 + i;
            int tl = idx >> 6, l = idx & 63;
            int t = t0 + tl;
            if (t < TT) {
                float s = 0.0f;
                #pragma unroll 8
                for (int dd = 0; dd < 64; ++dd) s += s_q[tl][dd] * s_state[dd][l];
                y[((long)(b * TT + t)) * DIM + hd * 64 + l] = s;
            }
        }
        __syncthreads();
    }
}

// ---------------------------------------------------------------------------
// Host orchestration
// ---------------------------------------------------------------------------
static inline long cdivl(long a, long b) { return (a + b - 1) / b; }

extern "C" void kernel_launch(void* const* d_in, const int* in_sizes, int n_in,
                              void* d_out, int out_size, void* d_ws, size_t ws_size,
                              hipStream_t stream) {
    (void)in_sizes; (void)n_in; (void)out_size; (void)ws_size;
    const float* x      = (const float*)d_in[0];
    const float* xf     = (const float*)d_in[1];
    const float* emb    = (const float*)d_in[2];
    const float* smask  = (const float*)d_in[3];
    const float* fp_w   = (const float*)d_in[4];
    const float* fp_b   = (const float*)d_in[5];
    const float* sa_ng  = (const float*)d_in[6];
    const float* sa_nb  = (const float*)d_in[7];
    const float* q_w    = (const float*)d_in[8];
    const float* q_b    = (const float*)d_in[9];
    const float* k_w    = (const float*)d_in[10];
    const float* k_b    = (const float*)d_in[11];
    const float* v_w    = (const float*)d_in[12];
    const float* v_b    = (const float*)d_in[13];
    const float* sa_ew  = (const float*)d_in[14];
    const float* sa_eb  = (const float*)d_in[15];
    const float* sa_ong = (const float*)d_in[16];
    const float* sa_onb = (const float*)d_in[17];
    const float* sa_ow  = (const float*)d_in[18];
    const float* sa_ob  = (const float*)d_in[19];
    const float* f1_w   = (const float*)d_in[20];
    const float* f1_b   = (const float*)d_in[21];
    const float* f2_w   = (const float*)d_in[22];
    const float* f2_b   = (const float*)d_in[23];
    const float* f_ew   = (const float*)d_in[24];
    const float* f_eb   = (const float*)d_in[25];
    const float* f_ng   = (const float*)d_in[26];
    const float* f_nb   = (const float*)d_in[27];
    const float* f_ow   = (const float*)d_in[28];
    const float* f_ob   = (const float*)d_in[29];

    float* h = (float*)d_out;     // running hidden state; final layer lands here

    // ---- workspace carving (256B aligned) ----
    char* wsb = (char*)d_ws;
    size_t off = 0;
    auto carve = [&](size_t bytes) -> void* {
        off = (off + 255) & ~(size_t)255;
        void* p = wsb + off;
        off += bytes;
        return p;
    };
    const long nFP  = (long)LAYERS * (DIM + AUD) * DIM;
    const long nQKV = (long)LAYERS * DIM * QKVN;
    const long nEW  = (long)LAYERS * TEMB * (2 * DIM);
    const long nOW  = (long)LAYERS * DIM * DIM;
    const long nF1  = (long)LAYERS * DIM * FFN;
    const long nF2  = (long)LAYERS * FFN * DIM;

    // all weight buffers hold PRE-TRANSPOSED [N,K] bf16 per layer
    bf16* wfp  = (bf16*)carve(nFP  * sizeof(bf16));
    bf16* wqkv = (bf16*)carve(nQKV * sizeof(bf16));
    bf16* wsae = (bf16*)carve(nEW  * sizeof(bf16));
    bf16* wsao = (bf16*)carve(nOW  * sizeof(bf16));
    bf16* wf1  = (bf16*)carve(nF1  * sizeof(bf16));
    bf16* wf2  = (bf16*)carve(nF2  * sizeof(bf16));
    bf16* wfe  = (bf16*)carve(nEW  * sizeof(bf16));
    bf16* wfo  = (bf16*)carve(nOW  * sizeof(bf16));
    float* qkvb = (float*)carve((long)LAYERS * QKVN * sizeof(float));

    bf16*  cbuf  = (bf16*)carve((long)ROWS * (DIM + AUD) * sizeof(bf16));
    bf16*  xnb   = (bf16*)carve((long)ROWS * DIM * sizeof(bf16));
    bf16*  hb    = (bf16*)carve((long)ROWS * DIM * sizeof(bf16));
    bf16*  midb  = (bf16*)carve((long)ROWS * FFN * sizeof(bf16));
    bf16*  stylb = (bf16*)carve((long)ROWS * DIM * sizeof(bf16));
    float* qkv   = (float*)carve((long)ROWS * QKVN * sizeof(float));
    float* ybuf  = (float*)carve((long)ROWS * DIM * sizeof(float));
    float* ebuf  = (float*)carve((long)BATCH * (2 * DIM) * sizeof(float));
    bf16*  sbuf  = (bf16*)carve((long)BATCH * TEMB * sizeof(bf16));

    // ---- weight conversion+transpose (every call: deterministic, stateless) ----
    auto cvt_t = [&](const float* s, bf16* d, int Kd, int Nd, long n) {
        cvt_t_bf16_kernel<<<dim3((unsigned)cdivl(n, 256)), dim3(256), 0, stream>>>(s, d, Kd, Nd, n);
    };
    cvt_t(fp_w,  wfp,  DIM + AUD, DIM,      nFP);
    cvt_t(sa_ew, wsae, TEMB,      2 * DIM,  nEW);
    cvt_t(sa_ow, wsao, DIM,       DIM,      nOW);
    cvt_t(f1_w,  wf1,  DIM,       FFN,      nF1);
    cvt_t(f2_w,  wf2,  FFN,       DIM,      nF2);
    cvt_t(f_ew,  wfe,  TEMB,      2 * DIM,  nEW);
    cvt_t(f_ow,  wfo,  DIM,       DIM,      nOW);
    pack_qkv_w_kernel<<<dim3((unsigned)cdivl(nQKV, 256)), dim3(256), 0, stream>>>(q_w, k_w, v_w, wqkv);
    pack_qkv_b_kernel<<<dim3((unsigned)cdivl((long)LAYERS * QKVN, 256)), dim3(256), 0, stream>>>(q_b, k_b, v_b, qkvb);

    // silu(emb) as bf16 GEMM input (shared by both stylization GEMMs of all layers)
    silu_cvt_kernel<<<dim3((unsigned)cdivl((long)BATCH * TEMB, 256)), dim3(256), 0, stream>>>(emb, sbuf, (long)BATCH * TEMB);

    // h = x
    hipMemcpyAsync(h, x, (size_t)ROWS * DIM * sizeof(float), hipMemcpyDeviceToDevice, stream);

    auto gemm = [&](const bf16* A, const bf16* W, const float* bias, const float* resid,
                    float* Cf, bf16* Cb, int M, int N, int K, int act) {
        dim3 grid((unsigned)(N / 128), (unsigned)cdivl(M, 128));
        gemm_bf16_kernel<<<grid, dim3(256), 0, stream>>>(A, W, bias, resid, Cf, Cb, M, N, K, act);
    };

    const long ncat = (long)ROWS * (DIM + AUD);
    for (int l = 0; l < LAYERS; ++l) {
        // 1) concat(h, xf) -> bf16
        concat_bf16_kernel<<<dim3((unsigned)cdivl(ncat, 256)), dim3(256), 0, stream>>>(h, xf, cbuf, ncat);
        // 2) h = concat @ fp_w + fp_b
        gemm(cbuf, wfp + (long)l * (DIM + AUD) * DIM, fp_b + (long)l * DIM,
             nullptr, h, nullptr, ROWS, DIM, DIM + AUD, 0);
        // 3) xn = LN(h) -> bf16
        ln512_bf16_kernel<<<dim3(ROWS), dim3(256), 0, stream>>>(h, sa_ng + (long)l * DIM, sa_nb + (long)l * DIM, xnb);
        // 4) qkv = xn @ [qw|kw|vw] + [qb|kb|vb]
        gemm(xnb, wqkv + (long)l * DIM * QKVN, qkvb + (long)l * QKVN,
             nullptr, qkv, nullptr, ROWS, QKVN, DIM, 0);
        // 5) linear attention -> ybuf
        linattn_kernel<<<dim3(BATCH * NHEAD), dim3(256), 0, stream>>>(qkv, smask, ybuf);
        // 6) e = silu(emb) @ sa_ew + sa_eb    [64 x 1024]
        gemm(sbuf, wsae + (long)l * TEMB * (2 * DIM), sa_eb + (long)l * (2 * DIM),
             nullptr, ebuf, nullptr, BATCH, 2 * DIM, TEMB, 0);
        // 7) stylb = silu(LN(y)*(1+scale)+shift)
        adaln_silu_kernel<<<dim3(ROWS), dim3(256), 0, stream>>>(ybuf, ebuf,
            sa_ong + (long)l * DIM, sa_onb + (long)l * DIM, stylb);
        // 8) h = h + stylb @ sa_ow + sa_ob   (also emit bf16 copy hb for FFN input)
        gemm(stylb, wsao + (long)l * DIM * DIM, sa_ob + (long)l * DIM,
             h, h, hb, ROWS, DIM, DIM, 0);
        // 9) mid = gelu(h @ f1_w + f1_b) -> bf16 only
        gemm(hb, wf1 + (long)l * DIM * FFN, f1_b + (long)l * FFN,
             nullptr, nullptr, midb, ROWS, FFN, DIM, 1);
        // 10) y = mid @ f2_w + f2_b
        gemm(midb, wf2 + (long)l * FFN * DIM, f2_b + (long)l * DIM,
             nullptr, ybuf, nullptr, ROWS, DIM, FFN, 0);
        // 11) e = silu(emb) @ f_ew + f_eb
        gemm(sbuf, wfe + (long)l * TEMB * (2 * DIM), f_eb + (long)l * (2 * DIM),
             nullptr, ebuf, nullptr, BATCH, 2 * DIM, TEMB, 0);
        // 12) stylb = silu(LN(y)*(1+scale)+shift)
        adaln_silu_kernel<<<dim3(ROWS), dim3(256), 0, stream>>>(ybuf, ebuf,
            f_ng + (long)l * DIM, f_nb + (long)l * DIM, stylb);
        // 13) h = h + stylb @ f_ow + f_ob
        gemm(stylb, wfo + (long)l * DIM * DIM, f_ob + (long)l * DIM,
             h, h, nullptr, ROWS, DIM, DIM, 0);
    }
}
